// CausalSelfAttention_63204738727965
// MI455X (gfx1250) — matrix operand order repaired and checked by reading, not compiled, not run on hardware
//
#include <hip/hip_runtime.h>
#include <hip/hip_bf16.h>

typedef __bf16 bf16;
typedef __attribute__((ext_vector_type(16))) __bf16 v16bf;
typedef __attribute__((ext_vector_type(8)))  float  v8f;
typedef __attribute__((ext_vector_type(4)))  int    v4i;

#if __has_builtin(__builtin_amdgcn_global_load_async_to_lds_b128)
#define HAVE_ASYNC_LDS 1
#else
#define HAVE_ASYNC_LDS 0
#endif

union Frag {
    v16bf v;
    uint4 u[2];
};

__device__ __forceinline__ v8f wmma_bf16(v16bf a, v16bf b, v8f c) {
    // D = A(16x32 bf16) * B(32x16 bf16) + C(16x16 f32)
    return __builtin_amdgcn_wmma_f32_16x16x32_bf16(
        /*neg_a=*/false, a, /*neg_b=*/false, b,
        /*c_mod=*/(short)0, c, /*reuse_a=*/false, /*reuse_b=*/false);
}

// 16B global -> LDS copy: async (ASYNCcnt-tracked, no VGPR round trip) when available.
__device__ __forceinline__ void cp_g2s_16B(bf16* ldst, const bf16* gsrc) {
#if HAVE_ASYNC_LDS
    __builtin_amdgcn_global_load_async_to_lds_b128(
        (__attribute__((address_space(1))) v4i*)gsrc,
        (__attribute__((address_space(3))) v4i*)ldst,
        0, 0);
#else
    *(uint4*)ldst = *(const uint4*)gsrc;
#endif
}

__device__ __forceinline__ void wait_async_lds() {
#if HAVE_ASYNC_LDS
#if __has_builtin(__builtin_amdgcn_s_wait_asynccnt)
    __builtin_amdgcn_s_wait_asynccnt(0);
#else
    asm volatile("s_wait_asynccnt 0x0" ::: "memory");
#endif
#endif
}

// ---------------------------------------------------------------------------
// Kernel 0: fp32 -> bf16 conversion (x, w_qkv, w_out). n must be multiple of 4.
// ---------------------------------------------------------------------------
__global__ __launch_bounds__(256) void cvt_f32_to_bf16(const float* __restrict__ src,
                                                       bf16* __restrict__ dst, int n) {
    int i = (blockIdx.x * blockDim.x + threadIdx.x) * 4;
    if (i < n) {
        float4 f = *(const float4*)(src + i);
        union { bf16 h[4]; uint2 u; } o;
        o.h[0] = (bf16)f.x; o.h[1] = (bf16)f.y; o.h[2] = (bf16)f.z; o.h[3] = (bf16)f.w;
        *(uint2*)(dst + i) = o.u;
    }
}

// ---------------------------------------------------------------------------
// bf16 GEMM: C[M,N] = A[M,K] @ B[N,K]^T + bias
// MODE 0: QKV epilogue -> scatter bf16 into Q[b,h,t,dh], K[b,h,t,dh], Vt[b,h,dh,t]
// MODE 1: plain fp32 store to out[M,N]
// Block tile 128x256, BK=32, 256 threads = 8 waves (2 in M x 4 in N),
// wave tile 64x64 -> 4x4 fragments (16 WMMA per K-step, 16 ds_load_b128).
// Double-buffered: next global K-slice is register-staged during compute.
// ---------------------------------------------------------------------------
template <int MODE>
__global__ __launch_bounds__(256) void gemm_bf16_kernel(
    const bf16* __restrict__ A,     // [M,K] row-major
    const bf16* __restrict__ Bw,    // [N,K] row-major
    const float* __restrict__ bias, // [N]
    bf16* __restrict__ q, bf16* __restrict__ k, bf16* __restrict__ vt, // MODE 0
    float* __restrict__ out,                                           // MODE 1
    int M, int N, int K)
{
    constexpr int BM = 128, BN = 256, BK = 32;
    constexpr int LDT = BK + 8; // padded LDS row stride (elements); 80B => 16B aligned
    __shared__ __align__(16) bf16 sA[BM * LDT];
    __shared__ __align__(16) bf16 sB[BN * LDT];

    const int tid  = threadIdx.x;
    const int wave = tid >> 5, lane = tid & 31;
    const int half = lane >> 4, l16 = lane & 15;
    const int wm = (wave & 1) * 64;   // 2 waves along M
    const int wn = (wave >> 1) * 64;  // 4 waves along N
    const int m0 = blockIdx.y * BM, n0 = blockIdx.x * BN;

    v8f acc[4][4] = {};

    // Global staging: A 128x32 -> 2 threads/row x 2 uint4; B 256x32 -> 1 thread/row x 4 uint4.
    const int arow = tid >> 1, acol = (tid & 1) * 16;
    const bf16* gA = A  + (size_t)(m0 + arow) * K + acol;
    const bf16* gB = Bw + (size_t)(n0 + tid) * K;

    uint4 ra0, ra1, rb0, rb1, rb2, rb3;
    ra0 = *(const uint4*)(gA);
    ra1 = *(const uint4*)(gA + 8);
    rb0 = *(const uint4*)(gB);
    rb1 = *(const uint4*)(gB + 8);
    rb2 = *(const uint4*)(gB + 16);
    rb3 = *(const uint4*)(gB + 24);

    for (int k0 = 0; k0 < K; k0 += BK) {
        // Commit staged registers to LDS.
        *(uint4*)(sA + arow * LDT + acol)     = ra0;
        *(uint4*)(sA + arow * LDT + acol + 8) = ra1;
        *(uint4*)(sB + tid * LDT)      = rb0;
        *(uint4*)(sB + tid * LDT + 8)  = rb1;
        *(uint4*)(sB + tid * LDT + 16) = rb2;
        *(uint4*)(sB + tid * LDT + 24) = rb3;
        __syncthreads();

        // Stage next K-slice from global while this slice is consumed.
        if (k0 + BK < K) {
            const bf16* nA = gA + k0 + BK;
            const bf16* nB = gB + k0 + BK;
            ra0 = *(const uint4*)(nA);
            ra1 = *(const uint4*)(nA + 8);
            rb0 = *(const uint4*)(nB);
            rb1 = *(const uint4*)(nB + 8);
            rb2 = *(const uint4*)(nB + 16);
            rb3 = *(const uint4*)(nB + 24);
            if (k0 + 2 * BK < K) {
                __builtin_prefetch(gA + k0 + 2 * BK, 0, 1);
                __builtin_prefetch(gB + k0 + 2 * BK, 0, 1);
            }
        }

        Frag af[4], bfr[4];
#pragma unroll
        for (int i = 0; i < 4; ++i) {
            const bf16* p = sA + (wm + 16 * i + l16) * LDT;
            af[i].u[0] = *(const uint4*)(p + 8 * half);        // K {0..7} / {8..15}
            af[i].u[1] = *(const uint4*)(p + 16 + 8 * half);   // K {16..23} / {24..31}
        }
#pragma unroll
        for (int j = 0; j < 4; ++j) {
            const bf16* p = sB + (wn + 16 * j + l16) * LDT;
            bfr[j].u[0] = *(const uint4*)(p + 8 * half);
            bfr[j].u[1] = *(const uint4*)(p + 16 + 8 * half);
        }
#pragma unroll
        for (int i = 0; i < 4; ++i)
#pragma unroll
            for (int j = 0; j < 4; ++j)
                acc[i][j] = wmma_bf16(af[i].v, bfr[j].v, acc[i][j]);
        __syncthreads();
    }

    // Epilogue. C fragment: lanes 0-15 rows r, lanes 16-31 rows r+8; col = l16.
#pragma unroll
    for (int i = 0; i < 4; ++i)
#pragma unroll
        for (int j = 0; j < 4; ++j)
#pragma unroll
            for (int r = 0; r < 8; ++r) {
                const int gm = m0 + wm + 16 * i + r + 8 * half;
                const int gn = n0 + wn + 16 * j + l16;
                const float v = acc[i][j][r] + bias[gn];
                if (MODE == 0) {
                    const int which = gn >> 10;     // 0=q 1=k 2=v
                    const int c  = gn & 1023;
                    const int h  = c >> 6, dh = c & 63;
                    const int bb = gm >> 11, t = gm & 2047;
                    const size_t bh = (size_t)bb * 16 + h;
                    if (which == 0)      q [(bh * 2048 + t) * 64 + dh] = (bf16)v;
                    else if (which == 1) k [(bh * 2048 + t) * 64 + dh] = (bf16)v;
                    else                 vt[(bh * 64 + dh) * 2048 + t] = (bf16)v;
                } else {
                    out[(size_t)gm * N + gn] = v;
                }
            }
}

// ---------------------------------------------------------------------------
// Flash-attention kernel.
// Grid: (B*H, T/128). 256 threads = 8 waves; wave owns 16 query rows.
// 128-key tiles staged in LDS (async-to-LDS when available); online softmax;
// causal mask only evaluated on the diagonal tile; P re-layout via per-wave LDS.
// ---------------------------------------------------------------------------
__global__ __launch_bounds__(256) void attn_kernel(
    const bf16* __restrict__ Q,   // [B*H, T, Dh]
    const bf16* __restrict__ Kt,  // [B*H, T, Dh]
    const bf16* __restrict__ Vt,  // [B*H, Dh, T]
    bf16* __restrict__ HO)        // [B*T, C] merged heads
{
    constexpr int T = 2048, Dh = 64, KT = 128;
    constexpr int LDK = 72;   // K-tile row stride (64+8), 144B aligned to 16B
    constexpr int LDV = 136;  // V-tile row stride (128+8), 272B aligned to 16B
    __shared__ __align__(16) bf16 sK[KT * LDK];        // 18 KB
    __shared__ __align__(16) bf16 sV[Dh * LDV];        // 17 KB
    __shared__ __align__(16) bf16 sP[8 * 16 * LDV];    // 34.8 KB (per-wave 16x128)

    const int bh = blockIdx.x, qt = blockIdx.y;
    const int b = bh >> 4, h = bh & 15;
    const int tid = threadIdx.x, wave = tid >> 5, lane = tid & 31;
    const int half = lane >> 4, l16 = lane & 15;

    const bf16* Qb = Q  + (size_t)bh * T * Dh;
    const bf16* Kb = Kt + (size_t)bh * T * Dh;
    const bf16* Vb = Vt + (size_t)bh * Dh * T;

    const int qrow0 = qt * 128 + wave * 16;

    // Load Q fragments once (A-operand layout) directly from global.
    Frag qa[2];
#pragma unroll
    for (int s = 0; s < 2; ++s) {
        const bf16* p = Qb + (size_t)(qrow0 + l16) * Dh + 32 * s;
        qa[s].u[0] = *(const uint4*)(p + 8 * half);
        qa[s].u[1] = *(const uint4*)(p + 16 + 8 * half);
    }

    float mrow[8], lrow[8];
#pragma unroll
    for (int r = 0; r < 8; ++r) { mrow[r] = -__builtin_huge_valf(); lrow[r] = 0.f; }
    v8f o[4] = {};

    bf16* pw = sP + wave * 16 * LDV; // per-wave P buffer: 16 x 128

    const int nkt = qt + 1; // 128-key tiles covering keys < qbase+128 (causal)
    for (int kt = 0; kt < nkt; ++kt) {
        const int tb = kt * KT;
        const bool diag = (kt == nkt - 1); // only tile that needs masking
        __syncthreads();
        // Stage K (128x64) and V^T (64x128): 1024 x 16B tasks each.
        for (int id = tid; id < 1024; id += 256) {
            const int krow = id >> 3, kc = (id & 7) * 8;
            cp_g2s_16B(sK + krow * LDK + kc, Kb + (size_t)(tb + krow) * Dh + kc);
            const int vrow = id >> 4, vc = (id & 15) * 8;
            cp_g2s_16B(sV + vrow * LDV + vc, Vb + (size_t)vrow * T + tb + vc);
        }
        wait_async_lds();
        __syncthreads();

        // S = Q @ K^T for 8 key fragments of 16.
        v8f sfr[8];
#pragma unroll
        for (int j = 0; j < 8; ++j) {
            v8f a = {};
#pragma unroll
            for (int s = 0; s < 2; ++s) {
                Frag kb;
                const bf16* p = sK + (16 * j + l16) * LDK + 32 * s;
                kb.u[0] = *(const uint4*)(p + 8 * half);
                kb.u[1] = *(const uint4*)(p + 16 + 8 * half);
                a = wmma_bf16(qa[s].v, kb.v, a);
            }
            sfr[j] = a;
        }

        // Scale, (diagonal-only) causal mask, online-softmax update.
        float alpha[8];
#pragma unroll
        for (int r = 0; r < 8; ++r) {
            const int qg = qrow0 + r + 8 * half;
            float mx = mrow[r];
#pragma unroll
            for (int j = 0; j < 8; ++j) {
                float v = sfr[j][r] * 0.125f;
                if (diag) {
                    const int kg = tb + 16 * j + l16;
                    v = (kg <= qg) ? v : -__builtin_huge_valf();
                }
                sfr[j][r] = v;
                mx = fmaxf(mx, v);
            }
#pragma unroll
            for (int m = 1; m < 16; m <<= 1) mx = fmaxf(mx, __shfl_xor(mx, m, 32));
            alpha[r] = (mrow[r] == mx) ? 1.f : __expf(mrow[r] - mx);
            lrow[r] *= alpha[r];
            mrow[r] = mx;
        }
#pragma unroll
        for (int d = 0; d < 4; ++d)
#pragma unroll
            for (int r = 0; r < 8; ++r) o[d][r] *= alpha[r];

        // p = exp(s - m); row-sum; store P (bf16) into per-wave LDS (C layout).
#pragma unroll
        for (int r = 0; r < 8; ++r) {
            float rs = 0.f;
#pragma unroll
            for (int j = 0; j < 8; ++j) {
                const float sv = sfr[j][r];
                const float pv = (sv == -__builtin_huge_valf()) ? 0.f : __expf(sv - mrow[r]);
                rs += pv;
                pw[(r + 8 * half) * LDV + 16 * j + l16] = (bf16)pv;
            }
#pragma unroll
            for (int m = 1; m < 16; m <<= 1) rs += __shfl_xor(rs, m, 32);
            lrow[r] += rs;
        }

        // Re-load P as A-operand fragments (wave-local LDS: DS ops stay in order).
        Frag pa[4];
#pragma unroll
        for (int s = 0; s < 4; ++s) {
            const bf16* p = pw + l16 * LDV + 32 * s;
            pa[s].u[0] = *(const uint4*)(p + 8 * half);
            pa[s].u[1] = *(const uint4*)(p + 16 + 8 * half);
        }
        // O += P @ V.  B-operand from V^T tile: row = dh, col = key.
#pragma unroll
        for (int d = 0; d < 4; ++d)
#pragma unroll
            for (int s = 0; s < 4; ++s) {
                Frag vb;
                const bf16* p = sV + (16 * d + l16) * LDV + 32 * s;
                vb.u[0] = *(const uint4*)(p + 8 * half);
                vb.u[1] = *(const uint4*)(p + 16 + 8 * half);
                o[d] = wmma_bf16(pa[s].v, vb.v, o[d]);
            }
    }

    // Normalize and store merged head output (bf16) at [b*T + t, h*64 + dh].
#pragma unroll
    for (int r = 0; r < 8; ++r) {
        const float inv = (lrow[r] > 0.f) ? 1.f / lrow[r] : 0.f;
        const int t = qrow0 + r + 8 * half;
        const size_t rowbase = ((size_t)b * 2048 + t) * 1024 + h * 64;
#pragma unroll
        for (int d = 0; d < 4; ++d)
            HO[rowbase + 16 * d + l16] = (bf16)(o[d][r] * inv);
    }
}

// ---------------------------------------------------------------------------
// Launch
// ---------------------------------------------------------------------------
extern "C" void kernel_launch(void* const* d_in, const int* in_sizes, int n_in,
                              void* d_out, int out_size, void* d_ws, size_t ws_size,
                              hipStream_t stream) {
    (void)in_sizes; (void)n_in; (void)out_size; (void)ws_size;
    constexpr int B = 4, T = 2048, C = 1024;
    constexpr int M = B * T;           // 8192
    const float* x     = (const float*)d_in[0];
    const float* w_qkv = (const float*)d_in[1];
    const float* b_qkv = (const float*)d_in[2];
    const float* w_out = (const float*)d_in[3];
    const float* b_out = (const float*)d_in[4];
    float* out = (float*)d_out;

    char* p = (char*)d_ws;
    bf16* Xb   = (bf16*)p; p += (size_t)M * C * 2;          // 16 MB
    bf16* Wqkv = (bf16*)p; p += (size_t)3 * C * C * 2;      //  6 MB
    bf16* Wout = (bf16*)p; p += (size_t)C * C * 2;          //  2 MB
    bf16* Qt   = (bf16*)p; p += (size_t)M * C * 2;          // 16 MB
    bf16* Kt   = (bf16*)p; p += (size_t)M * C * 2;          // 16 MB
    bf16* Vt   = (bf16*)p; p += (size_t)M * C * 2;          // 16 MB
    bf16* HO   = (bf16*)p; p += (size_t)M * C * 2;          // 16 MB

    // 1) fp32 -> bf16 conversions
    cvt_f32_to_bf16<<<(M * C) / 1024, 256, 0, stream>>>(x, Xb, M * C);
    cvt_f32_to_bf16<<<(3 * C * C) / 1024, 256, 0, stream>>>(w_qkv, Wqkv, 3 * C * C);
    cvt_f32_to_bf16<<<(C * C) / 1024, 256, 0, stream>>>(w_out, Wout, C * C);

    // 2) QKV projection + scatter (M=8192, N=3072, K=1024)
    gemm_bf16_kernel<0><<<dim3(3 * C / 256, M / 128), 256, 0, stream>>>(
        Xb, Wqkv, b_qkv, Qt, Kt, Vt, nullptr, M, 3 * C, C);

    // 3) flash attention
    attn_kernel<<<dim3(B * 16, T / 128), 256, 0, stream>>>(Qt, Kt, Vt, HO);

    // 4) output projection (M=8192, N=1024, K=1024), fp32 out + bias
    gemm_bf16_kernel<1><<<dim3(C / 256, M / 128), 256, 0, stream>>>(
        HO, Wout, b_out, nullptr, nullptr, nullptr, out, M, C, C);
}
